// SNNConvNet_20942260535415
// MI455X (gfx1250) — matrix operand order repaired and checked by reading, not compiled, not run on hardware
//
#include <hip/hip_runtime.h>
#include <hip/hip_bf16.h>
#include <stdint.h>

typedef __attribute__((ext_vector_type(16))) _Float16 v16h;
typedef __attribute__((ext_vector_type(8)))  _Float16 v8h;
typedef __attribute__((ext_vector_type(8)))  float    v8f;

// ---------------------------------------------------------------------------
// gfx1250 async global->LDS staging (ASYNCcnt path) via inline asm.
// GV mode: LDS address taken from the VDST VGPR; global addr = 64-bit VGPR pair.
// LDS byte address = low 32 bits of the generic pointer (flat->LDS rule:
// addrspacecast AS3->generic keeps AS3 value there).
// ---------------------------------------------------------------------------
__device__ __forceinline__ void stage16(const _Float16* g, _Float16* l) {
    uint32_t laddr = (uint32_t)(uintptr_t)l;
    uint64_t gaddr = (uint64_t)(uintptr_t)g;
    asm volatile("global_load_async_to_lds_b128 %0, %1, off"
                 :: "v"(laddr), "v"(gaddr)
                 : "memory");
}

__device__ __forceinline__ void async_join() {
    asm volatile("s_wait_asynccnt 0x0" ::: "memory");
}

// ---------------------------------------------------------------------------
// WMMA fragment loaders (CDNA5 wave32 VGPR layouts, 16-bit 16x16x32)
// A (16x32): lanes 0-15 = rows 0-15 with K {0..7,16..23}; lanes 16-31 same rows
//            with K {8..15,24..31}. Halves 0-7 = first run, 8-15 = second run.
// B (32x16): lane L -> column N=L%16; same K split as A (table order).
// C/D f32 : vgpr r -> M = r + 8*(L/16), N = L%16.
// ---------------------------------------------------------------------------
__device__ __forceinline__ v16h frag_a_row(const _Float16* rowp, int lane) {
    const int k0 = (lane & 16) ? 8 : 0;
    v8h lo = *(const v8h*)(rowp + k0);
    v8h hi = *(const v8h*)(rowp + k0 + 16);
    return __builtin_shufflevector(lo, hi, 0,1,2,3,4,5,6,7,8,9,10,11,12,13,14,15);
}
__device__ __forceinline__ v16h frag_b_row(const _Float16* rowp, int lane) {
    const int k0 = (lane & 16) ? 8 : 0;
    v8h lo = *(const v8h*)(rowp + k0);
    v8h hi = *(const v8h*)(rowp + k0 + 16);
    return __builtin_shufflevector(lo, hi, 0,1,2,3,4,5,6,7,8,9,10,11,12,13,14,15);
}
__device__ __forceinline__ v8f wmma_f16(v16h a, v16h b, v8f c) {
    return __builtin_amdgcn_wmma_f32_16x16x32_f16(false, a, false, b,
                                                  (short)0, c, false, false);
}

// ---------------------------------------------------------------------------
// Weight conversions
// ---------------------------------------------------------------------------
__global__ __launch_bounds__(256) void cvt_f32_to_f16_kernel(const float* __restrict__ src,
                                                             _Float16* __restrict__ dst, int n) {
    int i = blockIdx.x * 256 + threadIdx.x;
    if (i < n) dst[i] = (_Float16)src[i];
}

// conv2_w [64][32][3][3] (OIHW) -> w2t[j=(dy*3+dx)][n=64][ci=32] f16
__global__ __launch_bounds__(256) void cvt_w2t_kernel(const float* __restrict__ w2,
                                                      _Float16* __restrict__ w2t) {
    int t = blockIdx.x * 256 + threadIdx.x;   // 18432 total
    if (t >= 9 * 64 * 32) return;
    int ci = t & 31;
    int n  = (t >> 5) & 63;
    int j  = t >> 11;
    w2t[t] = (_Float16)w2[(n * 32 + ci) * 9 + j];
}

// ---------------------------------------------------------------------------
// conv1 (3->32, 3x3 same) + ReLU + maxpool2 fused, direct f32.
// Output pooled1 as NHWC f16: [512][16][16][32]  (c contiguous for conv2 GEMM)
// ---------------------------------------------------------------------------
__global__ __launch_bounds__(256) void conv1_pool_kernel(const float* __restrict__ x,
                                                         const float* __restrict__ w1,
                                                         const float* __restrict__ b1,
                                                         _Float16* __restrict__ out) {
    __shared__ float ws[32 * 27];
    __shared__ float bs[32];
    int t = threadIdx.x;
    for (int i = t; i < 32 * 27; i += 256) ws[i] = w1[i];
    if (t < 32) bs[t] = b1[t];
    __syncthreads();

    int g  = blockIdx.x * 256 + t;        // 4,194,304 total
    int c  = g & 31;
    int xp = (g >> 5) & 15;
    int yp = (g >> 9) & 15;
    int b  = g >> 13;
    const float* wc = &ws[c * 27];

    float best = 0.0f;                    // relu outputs are >= 0
    #pragma unroll
    for (int yy = 0; yy < 2; ++yy) {
        int oy = 2 * yp + yy;
        #pragma unroll
        for (int xx = 0; xx < 2; ++xx) {
            int ox = 2 * xp + xx;
            float acc = bs[c];
            for (int ci = 0; ci < 3; ++ci) {
                const float* xin = x + ((size_t)(b * 3 + ci)) * 1024;
                #pragma unroll
                for (int dy = 0; dy < 3; ++dy) {
                    int iy = oy + dy - 1;
                    if (iy < 0 || iy > 31) continue;
                    #pragma unroll
                    for (int dx = 0; dx < 3; ++dx) {
                        int ix = ox + dx - 1;
                        if (ix < 0 || ix > 31) continue;
                        acc += wc[ci * 9 + dy * 3 + dx] * xin[iy * 32 + ix];
                    }
                }
            }
            float r = acc > 0.f ? acc : 0.f;
            best = r > best ? r : best;
        }
    }
    out[g] = (_Float16)best;              // g == ((b*16+yp)*16+xp)*32 + c
}

// ---------------------------------------------------------------------------
// conv2 as implicit GEMM (M=512*16*16, N=64, K=9 taps x 32ch) via WMMA,
// fused bias + ReLU + maxpool2. Block = 256 thr = 8 waves (2M x 4N of 16x16),
// covers 32 pixels = one y-row-pair of one image -> pool in LDS.
// Output pooled2 NCHW f16 [512][64][8][8] == fc1 A matrix [512][4096].
// ---------------------------------------------------------------------------
__global__ __launch_bounds__(256) void conv2_gemm_pool_kernel(const _Float16* __restrict__ p1,
                                                              const _Float16* __restrict__ w2t,
                                                              const float* __restrict__ b2,
                                                              _Float16* __restrict__ p2) {
    __shared__ alignas(16) _Float16 As[32][32];   // [m local][ci]
    __shared__ alignas(16) _Float16 Bs[64][32];   // [n][ci]
    __shared__ float Cs[32][64];

    int t    = threadIdx.x;
    int lane = t & 31;
    int w    = t >> 5;
    int m0   = blockIdx.x * 32;                   // 4096 blocks
    int img  = m0 >> 8;
    int y0   = (m0 & 255) >> 4;                   // even

    int mOff = (w >> 2) * 16;
    int nOff = (w & 3) * 16;
    v8f acc = {};

    for (int j = 0; j < 9; ++j) {
        int dy = j / 3, dx = j % 3;
        if (j < 8)                                // prefetch next tap weights
            __builtin_prefetch(w2t + (j + 1) * 2048 + t * 8, 0, 3);
        if (t < 128) {                            // stage A: 32 rows x 64B
            int row = t >> 2;
            int ch  = t & 3;
            int m   = m0 + row;
            int y   = (m & 255) >> 4;
            int xx  = m & 15;
            int iy  = y + dy - 1;
            int ix  = xx + dx - 1;
            v8h val = {};
            if (iy >= 0 && iy < 16 && ix >= 0 && ix < 16)
                val = *(const v8h*)(p1 + ((((img * 16 + iy) * 16 + ix)) << 5) + ch * 8);
            *(v8h*)(&As[row][ch * 8]) = val;
        } else {                                  // stage B: 64 rows x 64B
            int tt = t - 128;
            int n  = tt >> 1;
            int ch = tt & 1;
            const v8h* src = (const v8h*)(w2t + (((j * 64 + n)) << 5) + ch * 16);
            v8h* dst = (v8h*)(&Bs[n][ch * 16]);
            dst[0] = src[0];
            dst[1] = src[1];
        }
        __syncthreads();
        v16h a = frag_a_row(&As[mOff + (lane & 15)][0], lane);
        v16h b = frag_b_row(&Bs[nOff + (lane & 15)][0], lane);
        acc = wmma_f16(a, b, acc);
        __syncthreads();
    }

    // bias + relu -> Cs
    {
        int n = nOff + (lane & 15);
        float bias = b2[n];
        #pragma unroll
        for (int r = 0; r < 8; ++r) {
            int m = mOff + r + ((lane & 16) >> 1);
            float v = acc[r] + bias;
            Cs[m][n] = v > 0.f ? v : 0.f;
        }
    }
    __syncthreads();

    // maxpool 2x2: local rows 0..15 = (y0, x), rows 16..31 = (y0+1, x)
    for (int o = t; o < 512; o += 256) {
        int c  = o & 63;
        int xp = o >> 6;
        float mx = fmaxf(fmaxf(Cs[2 * xp][c], Cs[2 * xp + 1][c]),
                         fmaxf(Cs[16 + 2 * xp][c], Cs[17 + 2 * xp][c]));
        p2[((((img * 64 + c) * 8 + (y0 >> 1))) << 3) + xp] = (_Float16)mx;
    }
}

// ---------------------------------------------------------------------------
// fc1: cur1[512,1000] = A[512,4096](f16) @ fc1_w^T (f16) + b.
// Block = 256 thr = 8 waves arranged 2Mw x 4Nw; each wave owns a 32x32 tile
// (2x2 WMMA accumulators -> 4 wmma per 2 A + 2 B fragment loads).
// Double-buffered LDS, async global->LDS staging, ONE barrier per K-chunk.
// Block tile 64M x 128N; grid 8 x 8; K loop 128 chunks of 32.
// ---------------------------------------------------------------------------
__global__ __launch_bounds__(256) void fc1_gemm_kernel(const _Float16* __restrict__ A,
                                                       const _Float16* __restrict__ Bw,
                                                       const float* __restrict__ bias,
                                                       float* __restrict__ cur1) {
    __shared__ alignas(16) _Float16 As[2][64][32];    //  8 KB
    __shared__ alignas(16) _Float16 Bs[2][128][32];   // 16 KB

    int t    = threadIdx.x;
    int lane = t & 31;
    int w    = t >> 5;
    int m0   = blockIdx.x * 64;                   // 8 blocks
    int n0   = blockIdx.y * 128;                  // 8 blocks (pad past 1000)
    int mW   = (w >> 2) * 32;
    int nW   = (w & 3) * 32;

    v8f acc00 = {}, acc01 = {}, acc10 = {}, acc11 = {};

    // staging assignments: A = 64 rows x 4 x 16B; B = 128 rows x 2 x 32B
    int arow = t >> 2, ach = t & 3;
    int brow = t >> 1, bch = t & 1;
    int gm  = m0 + arow;                          // < 512 always
    int gnB = n0 + brow; if (gnB > 999) gnB = 999;

    auto stage_chunk = [&](int buf, int kb) {
        stage16(A  + (size_t)gm  * 4096 + kb + ach * 8,       &As[buf][arow][ach * 8]);
        stage16(Bw + (size_t)gnB * 4096 + kb + bch * 16,      &Bs[buf][brow][bch * 16]);
        stage16(Bw + (size_t)gnB * 4096 + kb + bch * 16 + 8,  &Bs[buf][brow][bch * 16 + 8]);
    };

    stage_chunk(0, 0);
    for (int i = 0; i < 128; ++i) {
        int cur = i & 1;
        async_join();            // this thread's staging of buf[cur] complete
        __syncthreads();         // all staging visible; prev compute finished
        if (i + 1 < 128) stage_chunk(cur ^ 1, (i + 1) * 32);

        v16h a0 = frag_a_row(&As[cur][mW + (lane & 15)][0], lane);
        v16h a1 = frag_a_row(&As[cur][mW + 16 + (lane & 15)][0], lane);
        v16h b0 = frag_b_row(&Bs[cur][nW + (lane & 15)][0], lane);
        v16h b1 = frag_b_row(&Bs[cur][nW + 16 + (lane & 15)][0], lane);
        acc00 = wmma_f16(a0, b0, acc00);
        acc01 = wmma_f16(a0, b1, acc01);
        acc10 = wmma_f16(a1, b0, acc10);
        acc11 = wmma_f16(a1, b1, acc11);
    }

    int nc = n0 + nW + (lane & 15);
    int mb = m0 + mW + ((lane & 16) >> 1);
    auto store_acc = [&](v8f acc, int mBase, int nCol) {
        if (nCol < 1000) {
            #pragma unroll
            for (int r = 0; r < 8; ++r)
                cur1[(size_t)(mBase + r) * 1000 + nCol] = acc[r] + bias[nCol];
        }
    };
    store_acc(acc00, mb,      nc);
    store_acc(acc01, mb,      nc + 16);
    store_acc(acc10, mb + 16, nc);
    store_acc(acc11, mb + 16, nc + 16);
}

// ---------------------------------------------------------------------------
// LIF dynamics
// ---------------------------------------------------------------------------
__global__ __launch_bounds__(256) void zero_kernel(float* __restrict__ p, int n) {
    int i = blockIdx.x * 256 + threadIdx.x;
    if (i < n) p[i] = 0.f;
}

__global__ __launch_bounds__(256) void lif1_kernel(float* __restrict__ mem1,
                                                   const float* __restrict__ cur1,
                                                   float* __restrict__ spk1) {
    int i = blockIdx.x * 256 + threadIdx.x;       // 512000 exact
    float m = mem1[i];
    float reset = (m - 1.0f) > 0.f ? 1.0f : 0.f;  // subtract-reset from prev mem
    float mn = 0.95f * m + cur1[i] - reset;
    mem1[i] = mn;
    spk1[i] = (mn - 1.0f) > 0.f ? 1.f : 0.f;
}

// one wave per (b, j): cur2 = spk1[b,:] . fc2_w[j,:] + b; then LIF2 + outputs
__global__ __launch_bounds__(256) void lif2_kernel(const float* __restrict__ spk1,
                                                   const float* __restrict__ fc2w,
                                                   const float* __restrict__ fc2b,
                                                   float* __restrict__ mem2,
                                                   float* __restrict__ out, int step) {
    int wid  = (blockIdx.x * 256 + threadIdx.x) >> 5;   // 5120 waves exact
    int lane = threadIdx.x & 31;
    int b = wid / 10, j = wid % 10;
    const float* s    = spk1 + (size_t)b * 1000;
    const float* wrow = fc2w + (size_t)j * 1000;
    float sum = 0.f;
    for (int k = lane; k < 1000; k += 32) sum += s[k] * wrow[k];
    #pragma unroll
    for (int off = 16; off > 0; off >>= 1) sum += __shfl_xor(sum, off, 32);
    if (lane == 0) {
        float cur2  = sum + fc2b[j];
        float mp    = mem2[wid];
        float reset = (mp - 1.0f) > 0.f ? 1.0f : 0.f;
        float mn    = 0.95f * mp + cur2 - reset;
        mem2[wid]   = mn;
        out[step * 5120 + wid]         = (mn - 1.0f) > 0.f ? 1.f : 0.f;  // spk2_rec
        out[51200 + step * 5120 + wid] = mn;                             // mem2_rec
    }
}

// ---------------------------------------------------------------------------
extern "C" void kernel_launch(void* const* d_in, const int* in_sizes, int n_in,
                              void* d_out, int out_size, void* d_ws, size_t ws_size,
                              hipStream_t stream) {
    const float* x       = (const float*)d_in[0];
    const float* conv1_w = (const float*)d_in[1];
    const float* conv1_b = (const float*)d_in[2];
    const float* conv2_w = (const float*)d_in[3];
    const float* conv2_b = (const float*)d_in[4];
    const float* fc1_w   = (const float*)d_in[5];
    const float* fc1_b   = (const float*)d_in[6];
    const float* fc2_w   = (const float*)d_in[7];
    const float* fc2_b   = (const float*)d_in[8];
    float* out = (float*)d_out;

    // workspace carve (256B aligned)
    char* ws = (char*)d_ws;
    size_t off = 0;
    auto carve = [&](size_t bytes) -> char* {
        char* p = ws + off;
        off += (bytes + 255) & ~(size_t)255;
        return p;
    };
    _Float16* fc1w_h  = (_Float16*)carve(4096000u * 2);   // fc1_w f16
    _Float16* w2t_h   = (_Float16*)carve(18432u * 2);     // conv2_w transposed f16
    _Float16* pooled1 = (_Float16*)carve(4194304u * 2);   // [512][16][16][32] NHWC
    _Float16* pooled2 = (_Float16*)carve(2097152u * 2);   // [512][4096] (NCHW flat)
    float*    cur1    = (float*)carve(512000u * 4);
    float*    mem1    = (float*)carve(512000u * 4);
    float*    spk1    = (float*)carve(512000u * 4);
    float*    mem2    = (float*)carve(5120u * 4);
    (void)ws_size; (void)in_sizes; (void)n_in; (void)out_size;

    // 1) weight conversions
    cvt_f32_to_f16_kernel<<<16000, 256, 0, stream>>>(fc1_w, fc1w_h, 4096000);
    cvt_w2t_kernel<<<72, 256, 0, stream>>>(conv2_w, w2t_h);

    // 2) conv1 + relu + pool  -> pooled1 (NHWC f16)
    conv1_pool_kernel<<<16384, 256, 0, stream>>>(x, conv1_w, conv1_b, pooled1);

    // 3) conv2 implicit-GEMM (WMMA) + bias + relu + pool -> pooled2 (NCHW f16)
    conv2_gemm_pool_kernel<<<4096, 256, 0, stream>>>(pooled1, w2t_h, conv2_b, pooled2);

    // 4) fc1 GEMM (WMMA, double-buffered async staging) + bias -> cur1 f32
    fc1_gemm_kernel<<<dim3(8, 8), 256, 0, stream>>>(pooled2, fc1w_h, fc1_b, cur1);

    // 5) LIF state init (workspace is poisoned; must zero every call)
    zero_kernel<<<2000, 256, 0, stream>>>(mem1, 512000);
    zero_kernel<<<20, 256, 0, stream>>>(mem2, 5120);

    // 6) 10 sequential LIF timesteps
    for (int t = 0; t < 10; ++t) {
        lif1_kernel<<<2000, 256, 0, stream>>>(mem1, cur1, spk1);
        lif2_kernel<<<640, 256, 0, stream>>>(spk1, fc2_w, fc2_b, mem2, out, t);
    }
}
